// HandEncoder_7189775253535
// MI455X (gfx1250) — matrix operand order repaired and checked by reading, not compiled, not run on hardware
//
#include <hip/hip_runtime.h>

typedef __attribute__((ext_vector_type(16))) _Float16 v16h;
typedef __attribute__((ext_vector_type(8)))  _Float16 v8h;
typedef __attribute__((ext_vector_type(8)))  float    v8f;

#define BB 16
#define PP 64
#define KK 2048   // P * 32 contraction length (same for both layers)

// ---------------------------------------------------------------------------
// Stage A: per-pair hidden MLP  rel(3) -> 8 -> 16 -> 32, ReLU each layer.
// One thread per (b,j,k) pair; weights staged in LDS.
// Writes Hm[b][j][k*32+h] as f16 (WMMA A operand, row-major K).
// ---------------------------------------------------------------------------
__global__ void mlp_hidden_kernel(const float* __restrict__ coords,
                                  const float* __restrict__ W1, const float* __restrict__ b1,
                                  const float* __restrict__ W2, const float* __restrict__ b2,
                                  const float* __restrict__ W3, const float* __restrict__ b3,
                                  _Float16* __restrict__ Hm) {
  __shared__ float s[720];
  int tid = threadIdx.x;
  for (int i = tid; i < 24;  i += 256) s[i]     = W1[i];
  for (int i = tid; i < 8;   i += 256) s[24+i]  = b1[i];
  for (int i = tid; i < 128; i += 256) s[32+i]  = W2[i];
  for (int i = tid; i < 16;  i += 256) s[160+i] = b2[i];
  for (int i = tid; i < 512; i += 256) s[176+i] = W3[i];
  for (int i = tid; i < 32;  i += 256) s[688+i] = b3[i];
  __syncthreads();

  int t = blockIdx.x * 256 + tid;              // t = (b*64 + j)*64 + k
  int b = t >> 12, j = (t >> 6) & 63, k = t & 63;

  float r[3];
#pragma unroll
  for (int c = 0; c < 3; ++c)
    r[c] = coords[(b*PP + k)*3 + c] - coords[(b*PP + j)*3 + c];

  float h1[8];
#pragma unroll
  for (int o = 0; o < 8; ++o) {
    float a = s[24+o];
#pragma unroll
    for (int c = 0; c < 3; ++c) a += r[c] * s[c*8 + o];
    h1[o] = fmaxf(a, 0.f);
  }
  float h2[16];
#pragma unroll
  for (int o = 0; o < 16; ++o) {
    float a = s[160+o];
#pragma unroll
    for (int i = 0; i < 8; ++i) a += h1[i] * s[32 + i*16 + o];
    h2[o] = fmaxf(a, 0.f);
  }
#pragma unroll
  for (int o = 0; o < 32; ++o) {
    float a = s[688+o];
#pragma unroll
    for (int i = 0; i < 16; ++i) a += h2[i] * s[176 + i*32 + o];
    Hm[(size_t)t * 32 + o] = (_Float16)fmaxf(a, 0.f);
  }
}

// ---------------------------------------------------------------------------
// Stage B: G^T[b][d][k*32+h] = sum_c f[b,k,c] * W4[h, c*Cout + d]   (f16)
// One thread per (b,k,h).
// ---------------------------------------------------------------------------
template<int CIN, int COUT>
__global__ void gt_kernel(const float* __restrict__ feat, const float* __restrict__ W4,
                          _Float16* __restrict__ Gt) {
  int t = blockIdx.x * blockDim.x + threadIdx.x;  // (b*64 + k)*32 + h
  int b = t >> 11, k = (t >> 5) & 63, h = t & 31;
  float fr[CIN];
#pragma unroll
  for (int c = 0; c < CIN; ++c) fr[c] = feat[(b*PP + k)*CIN + c];
  const float* Wh = W4 + (size_t)h * CIN * COUT;
  int col = k*32 + h;
  for (int d = 0; d < COUT; ++d) {
    float a = 0.f;
#pragma unroll
    for (int c = 0; c < CIN; ++c) a += fr[c] * Wh[c*COUT + d];
    Gt[((size_t)(b*COUT + d)) * KK + col] = (_Float16)a;
  }
}

// bias_term[b][d] = sum_c (sum_k f[b,k,c]) * b4[c*Cout + d]
template<int CIN, int COUT>
__global__ void bias_kernel(const float* __restrict__ feat, const float* __restrict__ b4,
                            float* __restrict__ bias) {
  int t = blockIdx.x * blockDim.x + threadIdx.x;   // b*COUT + d
  if (t >= BB * COUT) return;
  int b = t / COUT, d = t % COUT;
  float acc = 0.f;
  for (int c = 0; c < CIN; ++c) {
    float s = 0.f;
    for (int k = 0; k < PP; ++k) s += feat[(b*PP + k)*CIN + c];
    acc += s * b4[c*COUT + d];
  }
  bias[t] = acc;
}

// ---------------------------------------------------------------------------
// Stage C: per-batch GEMM  out[j,d] = relu( H[j,:] . G[:,d] + bias[d] )
// M=64, N=COUT, K=2048. One wave per 16x16 tile, v_wmma_f32_16x16x32_f16.
// A lane layout: row = lane%16; elems 0-7 -> K = half*8+e, 8-15 -> 16+half*8+e.
// B lane layout (via Gt row = column d): elems 0-7 -> K = half*8+e, 8-15 -> 16+half*8+e.
// D: VGPR v -> row v + 8*half.
// ---------------------------------------------------------------------------
template<int COUT>
__global__ void pair_gemm_wmma(const _Float16* __restrict__ Hm,
                               const _Float16* __restrict__ Gt,
                               const float* __restrict__ bias,
                               float* __restrict__ outF) {
  constexpr int NT = COUT / 16;
  int tile = blockIdx.x;
  int b  = tile / (4 * NT);
  int r  = tile % (4 * NT);
  int tm = r / NT, tn = r % NT;
  int lane = threadIdx.x & 31;
  int half = lane >> 4, l16 = lane & 15;

  const _Float16* Ha = Hm + ((size_t)(b*PP   + tm*16 + l16)) * KK;
  const _Float16* Ga = Gt + ((size_t)(b*COUT + tn*16 + l16)) * KK;

  v8f acc = {};
#pragma unroll 4
  for (int kb = 0; kb < KK; kb += 32) {
    v8h a0 = *(const v8h*)(Ha + kb + half*8);
    v8h a1 = *(const v8h*)(Ha + kb + 16 + half*8);
    v16h A = __builtin_shufflevector(a0, a1, 0,1,2,3,4,5,6,7,8,9,10,11,12,13,14,15);
    v8h b0 = *(const v8h*)(Ga + kb + half*8);
    v8h b1 = *(const v8h*)(Ga + kb + 16 + half*8);
    v16h Bv = __builtin_shufflevector(b0, b1, 0,1,2,3,4,5,6,7,8,9,10,11,12,13,14,15);
    acc = __builtin_amdgcn_wmma_f32_16x16x32_f16(
        /*neg_a=*/false, A, /*neg_b=*/false, Bv,
        /*c_mod=*/(short)0, acc, /*reuse_a=*/false, /*reuse_b=*/false);
  }

  float bd = bias[b*COUT + tn*16 + l16];
#pragma unroll
  for (int v = 0; v < 8; ++v) {
    int row = tm*16 + v + 8*half;
    float val = fmaxf(acc[v] + bd, 0.f);
    outF[((size_t)(b*PP + row)) * COUT + tn*16 + l16] = val;
  }
}

// Final: out[b,d] = max_j feat2[b,j,d]
__global__ void max_kernel(const float* __restrict__ feat, float* __restrict__ out) {
  int t = blockIdx.x * blockDim.x + threadIdx.x;   // b*64 + d
  int b = t >> 6, d = t & 63;
  float m = -3.402823466e38f;
  for (int j = 0; j < PP; ++j) m = fmaxf(m, feat[(b*PP + j)*64 + d]);
  out[t] = m;
}

// ---------------------------------------------------------------------------
extern "C" void kernel_launch(void* const* d_in, const int* in_sizes, int n_in,
                              void* d_out, int out_size, void* d_ws, size_t ws_size,
                              hipStream_t stream) {
  const float* feature = (const float*)d_in[0];   // [16,64,16]
  const float* coords  = (const float*)d_in[1];   // [16,64,3]
  const float* p1[8]; const float* p2[8];
  for (int i = 0; i < 8; ++i) {
    p1[i] = (const float*)d_in[2 + i];            // W1,b1,W2,b2,W3,b3,W4,b4 (layer 1)
    p2[i] = (const float*)d_in[10 + i];           // same for layer 2
  }

  char* w = (char*)d_ws;
  _Float16* Hm   = (_Float16*)w; w += (size_t)BB*PP*KK*2;   // 4 MiB  H (f16)
  _Float16* Gt   = (_Float16*)w; w += (size_t)BB*64*KK*2;   // 4 MiB  G^T (f16), sized for Cout=64
  float*    feat1= (float*)w;    w += (size_t)BB*PP*32*4;   // layer-1 output feature
  float*    feat2= (float*)w;    w += (size_t)BB*PP*64*4;   // layer-2 output feature
  float*    biasb= (float*)w;    w += (size_t)BB*64*4;      // bias term

  // ---------------- layer 1: Cin=16, Cout=32 ----------------
  mlp_hidden_kernel<<<BB*PP*PP/256, 256, 0, stream>>>(
      coords, p1[0], p1[1], p1[2], p1[3], p1[4], p1[5], Hm);
  gt_kernel<16,32><<<BB*PP*32/256, 256, 0, stream>>>(feature, p1[6], Gt);
  bias_kernel<16,32><<<(BB*32 + 255)/256, 256, 0, stream>>>(feature, p1[7], biasb);
  pair_gemm_wmma<32><<<BB*4*2, 32, 0, stream>>>(Hm, Gt, biasb, feat1);

  // ---------------- layer 2: Cin=32, Cout=64 ----------------
  mlp_hidden_kernel<<<BB*PP*PP/256, 256, 0, stream>>>(
      coords, p2[0], p2[1], p2[2], p2[3], p2[4], p2[5], Hm);
  gt_kernel<32,64><<<BB*PP*32/256, 256, 0, stream>>>(feat1, p2[6], Gt);
  bias_kernel<32,64><<<(BB*64 + 255)/256, 256, 0, stream>>>(feat1, p2[7], biasb);
  pair_gemm_wmma<64><<<BB*4*4, 32, 0, stream>>>(Hm, Gt, biasb, feat2);

  // ---------------- max over nodes ----------------
  max_kernel<<<BB*64/256, 256, 0, stream>>>(feat2, (float*)d_out);
}
